// CorrespondenceEngine_29703993819774
// MI455X (gfx1250) — matrix operand order repaired and checked by reading, not compiled, not run on hardware
//
#include <hip/hip_runtime.h>
#include <hip/hip_bf16.h>
#include <math.h>

typedef __attribute__((ext_vector_type(16))) _Float16 v16h;
typedef __attribute__((ext_vector_type(8)))  float    v8f;

#define B_      2
#define C_      128
#define H0_     384
#define W0_     512
#define H1_     192
#define W1_     256
#define N0_     (H0_*W0_)     /* 196608 */
#define N1_     (H1_*W1_)     /* 49152  */
#define NTOT_   (N0_+N1_)     /* 245760 */
#define K_      512
#define KM_     128
#define NBINS_  1024
#define CANDMAX_ 4096

// ---------------------------------------------------------------------------
// Kernel 1: masked NMS scores.  scores[img][b][j] = sqrt(rep*rel) if (rep is
// a 3x3 local max && mean >= 0.7) else -1.  j < N0_ -> scale0, else scale1.
// ---------------------------------------------------------------------------
__global__ void score_kernel(const float* __restrict__ rep1_s0, const float* __restrict__ rel1_s0,
                             const float* __restrict__ rep1_s1, const float* __restrict__ rel1_s1,
                             const float* __restrict__ rep2_s0, const float* __restrict__ rel2_s0,
                             const float* __restrict__ rep2_s1, const float* __restrict__ rel2_s1,
                             float* __restrict__ scores) {
    long gid = (long)blockIdx.x * blockDim.x + threadIdx.x;
    const long total = 2L * B_ * NTOT_;
    if (gid >= total) return;
    int img = (int)(gid / (B_ * (long)NTOT_));
    int rem = (int)(gid % (B_ * (long)NTOT_));
    int b = rem / NTOT_;
    int j = rem % NTOT_;
    const float* rep; const float* rel; int H, Wd, p;
    if (j < N0_) { H = H0_; Wd = W0_; p = j;
        rep = img ? rep2_s0 : rep1_s0; rel = img ? rel2_s0 : rel1_s0; }
    else { H = H1_; Wd = W1_; p = j - N0_;
        rep = img ? rep2_s1 : rep1_s1; rel = img ? rel2_s1 : rel1_s1; }
    const float* rb = rep + (long)b * H * Wd;
    int r = p / Wd, c = p % Wd;
    float v = rb[p];
    float m = v;
    #pragma unroll
    for (int dr = -1; dr <= 1; ++dr)
        for (int dc = -1; dc <= 1; ++dc) {
            int rr = r + dr, cc = c + dc;
            if (rr >= 0 && rr < H && cc >= 0 && cc < Wd)
                m = fmaxf(m, rb[rr * Wd + cc]);
        }
    float mean = sqrtf(fmaxf(v * rel[(long)b * H * Wd + p], 0.0f));
    scores[gid] = (v == m && mean >= 0.7f) ? mean : -1.0f;
}

// ---------------------------------------------------------------------------
// Kernel 2: per-row (img,b) top-512.  Histogram over [0.7,1.0] -> threshold
// bin -> collect candidates -> LDS bitonic sort (desc score, asc idx).
// ---------------------------------------------------------------------------
__global__ __launch_bounds__(1024) void topk_kernel(const float* __restrict__ scores,
                                                    int* __restrict__ topk_idx,
                                                    float* __restrict__ topk_sc) {
    int row = blockIdx.x;                      // img*B_ + b, 4 rows total
    const float* s = scores + (long)row * NTOT_;
    __shared__ int hist[NBINS_];
    __shared__ int s_tbin, s_nc;
    __shared__ float csc[CANDMAX_];
    __shared__ int cidx[CANDMAX_];
    int tid = threadIdx.x;
    for (int i = tid; i < NBINS_; i += 1024) hist[i] = 0;
    if (tid == 0) s_nc = 0;
    __syncthreads();
    for (int j = tid; j < NTOT_; j += 1024) {
        float v = s[j];
        if (v > 0.0f) {
            int bin = (int)((v - 0.7f) * (NBINS_ / 0.3f));
            bin = bin < 0 ? 0 : (bin > NBINS_ - 1 ? NBINS_ - 1 : bin);
            atomicAdd(&hist[bin], 1);
        }
    }
    __syncthreads();
    if (tid == 0) {
        int total = 0, tb = 0;
        for (int b = NBINS_ - 1; b >= 0; --b) {
            total += hist[b];
            if (total >= K_) { tb = b; break; }
        }
        s_tbin = tb;
    }
    __syncthreads();
    int tbin = s_tbin;
    for (int j = tid; j < NTOT_; j += 1024) {
        float v = s[j];
        if (v > 0.0f) {
            int bin = (int)((v - 0.7f) * (NBINS_ / 0.3f));
            bin = bin < 0 ? 0 : (bin > NBINS_ - 1 ? NBINS_ - 1 : bin);
            if (bin >= tbin) {
                int slot = atomicAdd(&s_nc, 1);
                if (slot < CANDMAX_) { csc[slot] = v; cidx[slot] = j; }
            }
        }
    }
    __syncthreads();
    int nc = s_nc; if (nc > CANDMAX_) nc = CANDMAX_;
    for (int i = tid; i < CANDMAX_; i += 1024)
        if (i >= nc) { csc[i] = -2.0f; cidx[i] = 0x7FFFFFFF; }
    __syncthreads();
    // bitonic sort: "less" == comes-first == (higher score) or (==, lower idx)
    for (int ksz = 2; ksz <= CANDMAX_; ksz <<= 1) {
        for (int jj = ksz >> 1; jj > 0; jj >>= 1) {
            for (int t = tid; t < CANDMAX_; t += 1024) {
                int ixj = t ^ jj;
                if (ixj > t) {
                    bool up = ((t & ksz) == 0);
                    float ka = csc[t], kb = csc[ixj];
                    int ia = cidx[t], ib = cidx[ixj];
                    bool b_first = (kb > ka) || (kb == ka && ib < ia);  // elem[ixj] "less"
                    bool a_first = (ka > kb) || (ka == kb && ia < ib);
                    bool sw = up ? b_first : a_first;
                    if (sw) { csc[t] = kb; csc[ixj] = ka; cidx[t] = ib; cidx[ixj] = ia; }
                }
            }
            __syncthreads();
        }
    }
    for (int k = tid; k < K_; k += 1024) {
        float v = csc[k]; int id = cidx[k];
        if (v <= -1.5f) { v = -1.0f; id = (k - nc) < 0 ? 0 : (k - nc); }  // pad like jax -1 fill
        topk_sc[row * K_ + k] = v;
        topk_idx[row * K_ + k] = id;
    }
}

// ---------------------------------------------------------------------------
// Kernel 3: gather locations + descriptors (scaled by score) -> f16
// Dh[row][k][c].  One block per (row,k), one thread per channel.
// ---------------------------------------------------------------------------
__global__ __launch_bounds__(128) void gather_kernel(
        const float* __restrict__ d1_s0, const float* __restrict__ d1_s1,
        const float* __restrict__ d2_s0, const float* __restrict__ d2_s1,
        const int* __restrict__ topk_idx, const float* __restrict__ topk_sc,
        float* __restrict__ loc, _Float16* __restrict__ Dh) {
    int row = blockIdx.x / K_;      // img*B_ + b
    int k = blockIdx.x % K_;
    int img = row / B_;
    int b = row % B_;
    int j = topk_idx[row * K_ + k];
    if (j < 0) j = 0; if (j >= NTOT_) j = 0;
    float sc = topk_sc[row * K_ + k];
    const float* ds; int HW, pos; float lx, ly;
    if (j < N0_) { ds = img ? d2_s0 : d1_s0; HW = N0_; pos = j;
        lx = (float)(j / W0_); ly = (float)(j % W0_); }
    else { int jj = j - N0_; ds = img ? d2_s1 : d1_s1; HW = N1_; pos = jj;
        lx = 2.0f * (float)(jj / W1_); ly = 2.0f * (float)(jj % W1_); }
    int c = threadIdx.x;
    float v = ds[((long)b * C_ + c) * HW + pos] * sc;
    Dh[((long)row * K_ + k) * C_ + c] = (_Float16)v;
    if (c == 0) {
        loc[((row * 2) + 0) * K_ + k] = lx;
        loc[((row * 2) + 1) * K_ + k] = ly;
    }
}

// ---------------------------------------------------------------------------
// Kernel 4: projection Q = D1*Wq^T, K = D2*Wk^T via V_WMMA_F32_16X16X32_F16.
// One wave per 16x16 output tile.  A frag layout: M=lane%16, K per ISA table;
// B frag: N=lane%16, K per ISA table (same as A).
// ---------------------------------------------------------------------------
__global__ __launch_bounds__(32) void proj_kernel(const _Float16* __restrict__ Dh,
                                                  const float* __restrict__ Wq,
                                                  const float* __restrict__ Wk,
                                                  _Float16* __restrict__ QKh) {
    int id = blockIdx.x;
    int tn = id & 7;  id >>= 3;    // 8 N-tiles (output channel r)
    int tm = id & 31; id >>= 5;    // 32 M-tiles (keypoints)
    int b  = id & (B_ - 1); id >>= 1;
    int pk = id;                   // 0 -> Q (img0 desc, Wq), 1 -> K (img1 desc, Wk)
    const _Float16* D = Dh + ((long)(pk * B_ + b)) * K_ * C_;
    const float* W = pk ? Wk : Wq;
    int lane = threadIdx.x;
    int g = lane >> 4, l = lane & 15;
    v8f acc = {};
    #pragma unroll
    for (int kb = 0; kb < C_; kb += 32) {
        const _Float16* arow = D + (long)(tm * 16 + l) * C_ + kb;
        const float*    wrow = W + (long)(tn * 16 + l) * C_ + kb;  // B elem (K,N=r)=W[r][K]
        v16h a, bf;
        #pragma unroll
        for (int p = 0; p < 16; ++p) {
            int ka = (p < 8) ? (g * 8 + p) : (16 + g * 8 + (p - 8));
            a[p]  = arow[ka];
            bf[p] = (_Float16)wrow[ka];
        }
        acc = __builtin_amdgcn_wmma_f32_16x16x32_f16(false, a, false, bf,
                                                     (short)0, acc, false, false);
    }
    _Float16* out = QKh + ((long)(pk * B_ + b)) * K_ * C_;
    #pragma unroll
    for (int v = 0; v < 8; ++v)
        out[(long)(tm * 16 + v + 8 * g) * C_ + tn * 16 + l] = (_Float16)acc[v];
}

// ---------------------------------------------------------------------------
// Kernel 5: rel[b][n][m] = sum_r Q[b][n][r] * K[b][m][r]  (512x128 x 128x512)
// ---------------------------------------------------------------------------
__global__ __launch_bounds__(32) void rel_kernel(const _Float16* __restrict__ QKh,
                                                 float* __restrict__ rel) {
    int id = blockIdx.x;
    int tn = id & 31; id >>= 5;    // 32 N-tiles over m (image2 kps)
    int tm = id & 31; id >>= 5;    // 32 M-tiles over n (image1 kps)
    int b = id;
    const _Float16* Q  = QKh + ((long)(0 * B_ + b)) * K_ * C_;
    const _Float16* Kd = QKh + ((long)(1 * B_ + b)) * K_ * C_;
    int lane = threadIdx.x;
    int g = lane >> 4, l = lane & 15;
    v8f acc = {};
    #pragma unroll
    for (int kb = 0; kb < C_; kb += 32) {
        const _Float16* arow = Q  + (long)(tm * 16 + l) * C_ + kb;
        const _Float16* brow = Kd + (long)(tn * 16 + l) * C_ + kb;  // B elem (K=r,N=m)=K[m][r]
        __builtin_prefetch(arow + 32, 0, 1);
        __builtin_prefetch(brow + 32, 0, 1);
        v16h a, bf;
        #pragma unroll
        for (int p = 0; p < 16; ++p) {
            int ka = (p < 8) ? (g * 8 + p) : (16 + g * 8 + (p - 8));
            a[p]  = arow[ka];
            bf[p] = brow[ka];
        }
        acc = __builtin_amdgcn_wmma_f32_16x16x32_f16(false, a, false, bf,
                                                     (short)0, acc, false, false);
    }
    float* R = rel + (long)b * K_ * K_;
    #pragma unroll
    for (int v = 0; v < 8; ++v)
        R[(long)(tm * 16 + v + 8 * g) * K_ + tn * 16 + l] = acc[v];
}

// ---------------------------------------------------------------------------
// Kernel 6: per (b,n): top-2 of rel row, acos ratio, softmax(rel*512),
// y = attn * (Wv @ loc2).  One 256-thread block per row.
// ---------------------------------------------------------------------------
__global__ __launch_bounds__(256) void attn_kernel(const float* __restrict__ rel,
                                                   const float* __restrict__ loc,
                                                   const float* __restrict__ Wv,
                                                   float* __restrict__ corr,
                                                   float* __restrict__ ratios) {
    int n = blockIdx.x % K_;
    int b = blockIdx.x / K_;
    const float* R = rel + ((long)b * K_ + n) * K_;
    const float* l2x = loc + (((1 * B_ + b) * 2) + 0) * K_;
    const float* l2y = loc + (((1 * B_ + b) * 2) + 1) * K_;
    int tid = threadIdx.x;
    float r0 = R[tid], r1 = R[tid + 256];
    __shared__ float sh1[256], sh2[256];
    sh1[tid] = fmaxf(r0, r1);
    sh2[tid] = fminf(r0, r1);
    __syncthreads();
    for (int s = 128; s > 0; s >>= 1) {
        if (tid < s) {
            float a1 = sh1[tid], a2 = sh2[tid], b1 = sh1[tid + s], b2 = sh2[tid + s];
            sh1[tid] = fmaxf(a1, b1);
            sh2[tid] = fmaxf(fminf(a1, b1), fmaxf(a2, b2));
        }
        __syncthreads();
    }
    float best1 = sh1[0], best2 = sh2[0];
    __syncthreads();
    float wv00 = Wv[0], wv01 = Wv[1], wv10 = Wv[2], wv11 = Wv[3];
    float e0 = expf((r0 - best1) * 512.0f);
    float e1 = expf((r1 - best1) * 512.0f);
    float vx0 = wv00 * l2x[tid] + wv01 * l2y[tid];
    float vx1 = wv00 * l2x[tid + 256] + wv01 * l2y[tid + 256];
    float vy0 = wv10 * l2x[tid] + wv11 * l2y[tid];
    float vy1 = wv10 * l2x[tid + 256] + wv11 * l2y[tid + 256];
    __shared__ float rs0[256], rs1[256], rs2[256];
    rs0[tid] = e0 + e1;
    rs1[tid] = e0 * vx0 + e1 * vx1;
    rs2[tid] = e0 * vy0 + e1 * vy1;
    __syncthreads();
    for (int s = 128; s > 0; s >>= 1) {
        if (tid < s) {
            rs0[tid] += rs0[tid + s];
            rs1[tid] += rs1[tid + s];
            rs2[tid] += rs2[tid + s];
        }
        __syncthreads();
    }
    if (tid == 0) {
        float inv = 1.0f / rs0[0];
        corr[((long)b * 2 + 0) * K_ + n] = rs1[0] * inv;
        corr[((long)b * 2 + 1) * K_ + n] = rs2[0] * inv;
        float c1 = fminf(fmaxf(best1, -1.0f), 1.0f);
        float c2 = fminf(fmaxf(best2, -1.0f), 1.0f);
        float rat = acosf(c1) / acosf(c2);
        if (!(rat == rat)) rat = 1e30f;
        ratios[(long)b * K_ + n] = rat;
    }
}

// ---------------------------------------------------------------------------
// Kernel 7: per b, sort 512 ratios ascending (tie: lower idx) in LDS, take
// first 128, emit matches [loc1x, loc1y, corr0, corr1] -> out [B,4,128].
// ---------------------------------------------------------------------------
__global__ __launch_bounds__(256) void final_kernel(const float* __restrict__ ratios,
                                                    const float* __restrict__ loc,
                                                    const float* __restrict__ corr,
                                                    float* __restrict__ out) {
    int b = blockIdx.x;
    __shared__ float rk[K_];
    __shared__ int ri[K_];
    int tid = threadIdx.x;
    for (int i = tid; i < K_; i += 256) {
        float v = ratios[(long)b * K_ + i];
        if (!(v == v)) v = 1e30f;
        rk[i] = v; ri[i] = i;
    }
    __syncthreads();
    for (int ksz = 2; ksz <= K_; ksz <<= 1) {
        for (int j = ksz >> 1; j > 0; j >>= 1) {
            for (int t = tid; t < K_; t += 256) {
                int ixj = t ^ j;
                if (ixj > t) {
                    bool up = ((t & ksz) == 0);
                    float ka = rk[t], kb = rk[ixj];
                    int ia = ri[t], ib = ri[ixj];
                    bool b_less = (kb < ka) || (kb == ka && ib < ia);
                    bool a_less = (ka < kb) || (ka == kb && ia < ib);
                    bool sw = up ? b_less : a_less;
                    if (sw) { rk[t] = kb; rk[ixj] = ka; ri[t] = ib; ri[ixj] = ia; }
                }
            }
            __syncthreads();
        }
    }
    const float* l1x = loc + (((0 * B_ + b) * 2) + 0) * K_;
    const float* l1y = loc + (((0 * B_ + b) * 2) + 1) * K_;
    const float* c0 = corr + ((long)b * 2 + 0) * K_;
    const float* c1 = corr + ((long)b * 2 + 1) * K_;
    if (tid < KM_) {
        int n = ri[tid];
        float* ob = out + (long)b * 4 * KM_;
        ob[0 * KM_ + tid] = l1x[n];
        ob[1 * KM_ + tid] = l1y[n];
        ob[2 * KM_ + tid] = c0[n];
        ob[3 * KM_ + tid] = c1[n];
    }
}

// ---------------------------------------------------------------------------
extern "C" void kernel_launch(void* const* d_in, const int* in_sizes, int n_in,
                              void* d_out, int out_size, void* d_ws, size_t ws_size,
                              hipStream_t stream) {
    (void)in_sizes; (void)n_in; (void)out_size; (void)ws_size;
    const float* rep1_s0  = (const float*)d_in[0];
    const float* rel1_s0  = (const float*)d_in[1];
    const float* desc1_s0 = (const float*)d_in[2];
    const float* rep1_s1  = (const float*)d_in[3];
    const float* rel1_s1  = (const float*)d_in[4];
    const float* desc1_s1 = (const float*)d_in[5];
    const float* rep2_s0  = (const float*)d_in[6];
    const float* rel2_s0  = (const float*)d_in[7];
    const float* desc2_s0 = (const float*)d_in[8];
    const float* rep2_s1  = (const float*)d_in[9];
    const float* rel2_s1  = (const float*)d_in[10];
    const float* desc2_s1 = (const float*)d_in[11];
    const float* Wq = (const float*)d_in[12];
    const float* Wk = (const float*)d_in[13];
    const float* Wv = (const float*)d_in[14];

    char* ws = (char*)d_ws;
    size_t off = 0;
    auto carve = [&](size_t bytes) -> void* {
        off = (off + 255) & ~(size_t)255;
        void* p = ws + off;
        off += bytes;
        return p;
    };
    float*    scores  = (float*)   carve((size_t)2 * B_ * NTOT_ * sizeof(float));
    int*      tk_idx  = (int*)     carve((size_t)2 * B_ * K_ * sizeof(int));
    float*    tk_sc   = (float*)   carve((size_t)2 * B_ * K_ * sizeof(float));
    float*    loc     = (float*)   carve((size_t)2 * B_ * 2 * K_ * sizeof(float));
    _Float16* Dh      = (_Float16*)carve((size_t)2 * B_ * K_ * C_ * sizeof(_Float16));
    _Float16* QKh     = (_Float16*)carve((size_t)2 * B_ * K_ * C_ * sizeof(_Float16));
    float*    rel     = (float*)   carve((size_t)B_ * K_ * K_ * sizeof(float));
    float*    corr    = (float*)   carve((size_t)B_ * 2 * K_ * sizeof(float));
    float*    ratio   = (float*)   carve((size_t)B_ * K_ * sizeof(float));

    const long total = 2L * B_ * NTOT_;
    score_kernel<<<(int)((total + 255) / 256), 256, 0, stream>>>(
        rep1_s0, rel1_s0, rep1_s1, rel1_s1, rep2_s0, rel2_s0, rep2_s1, rel2_s1, scores);
    topk_kernel<<<2 * B_, 1024, 0, stream>>>(scores, tk_idx, tk_sc);
    gather_kernel<<<2 * B_ * K_, 128, 0, stream>>>(
        desc1_s0, desc1_s1, desc2_s0, desc2_s1, tk_idx, tk_sc, loc, Dh);
    proj_kernel<<<2 * B_ * 32 * 8, 32, 0, stream>>>(Dh, Wq, Wk, QKh);
    rel_kernel<<<B_ * 32 * 32, 32, 0, stream>>>(QKh, rel);
    attn_kernel<<<B_ * K_, 256, 0, stream>>>(rel, loc, Wv, corr, ratio);
    final_kernel<<<B_, 256, 0, stream>>>(ratio, loc, corr, (float*)d_out);
}
